// BiLSTM_M_61615600828569
// MI455X (gfx1250) — matrix operand order repaired and checked by reading, not compiled, not run on hardware
//
#include <hip/hip_runtime.h>
#include <hip/hip_bf16.h>

// Problem constants (from the reference)
#define B_      16
#define S_      1024
#define D_      768
#define NS_     256
#define MAXW_   8
#define V_      128
#define C_      6
#define R_      512
#define REL_    97
#define HID_    384
#define DIS_    20
#define NEG_    (-1e30f)
#define FIN_    (D_ + DIS_)         // 788
#define INPROJ_ (2 * FIN_ + D_)     // 2344
#define KP1_    2368                // INPROJ_ padded to 74*32 for WMMA K-steps
#define NP2_    112                 // REL_ padded to 7*16
#define M_      (B_ * R_)           // 8192 rows in both GEMMs

// GEMM1 macro-tile
#define BM_     128
#define BN_     64
#define BK_     32

typedef __attribute__((ext_vector_type(16))) _Float16 v16h;
typedef __attribute__((ext_vector_type(8)))  _Float16 v8h;
typedef __attribute__((ext_vector_type(8)))  float    v8f;

// ---------------------------------------------------------------------------
// CDNA5 async copy helpers (ASYNCcnt-tracked, LDS-direct)
// ---------------------------------------------------------------------------
__device__ __forceinline__ void async_ld_b128(uint32_t lds_off, const void* gptr) {
    asm volatile("global_load_async_to_lds_b128 %0, %1, off"
                 :: "v"(lds_off), "v"(gptr) : "memory");
}
__device__ __forceinline__ void wait_async0() {
    asm volatile("s_wait_asynccnt 0x0" ::: "memory");
}

// ---------------------------------------------------------------------------
// Pack W1 (2344 x 384, row-major) -> W1T f16 (384 x 2368, K-major, zero pad K)
// ---------------------------------------------------------------------------
__global__ void __launch_bounds__(256) pack_w1t_kernel(const float* __restrict__ W1,
                                                       _Float16* __restrict__ W1T) {
    int idx = blockIdx.x * 256 + threadIdx.x;
    if (idx >= HID_ * KP1_) return;
    int n = idx / KP1_;
    int k = idx % KP1_;
    float v = (k < INPROJ_) ? W1[(size_t)k * HID_ + n] : 0.0f;
    W1T[idx] = (_Float16)v;
}

// ---------------------------------------------------------------------------
// Pack W2 (384 x 97) -> W2T f16 (112 x 384, K-major, zero pad N)
// ---------------------------------------------------------------------------
__global__ void __launch_bounds__(256) pack_w2t_kernel(const float* __restrict__ W2,
                                                       _Float16* __restrict__ W2T) {
    int idx = blockIdx.x * 256 + threadIdx.x;
    if (idx >= NP2_ * HID_) return;
    int n = idx / HID_;
    int k = idx % HID_;
    float v = (n < REL_) ? W2[(size_t)k * REL_ + n] : 0.0f;
    W2T[idx] = (_Float16)v;
}

// ---------------------------------------------------------------------------
// Span max-pool: one block per (b, ns); masked max over up to 8 tokens.
// ---------------------------------------------------------------------------
__global__ void __launch_bounds__(256) span_max_kernel(const float* __restrict__ sr,
                                                       const int* __restrict__ esi,
                                                       float* __restrict__ spanEmb) {
    const int bs = blockIdx.x;                // b*NS + ns
    const int b = bs / NS_;
    const int start = esi[bs * 2 + 0];
    const int end   = esi[bs * 2 + 1];
    const float* base = sr + (size_t)b * S_ * D_;
    for (int d = threadIdx.x; d < D_; d += 256) {
        float m = NEG_;
        #pragma unroll
        for (int w = 0; w < MAXW_; ++w) {
            int pos = start + w;
            if (pos <= end) {
                int pc = pos < (S_ - 1) ? pos : (S_ - 1);
                float v = base[(size_t)pc * D_ + d];
                m = v > m ? v : m;
            }
        }
        spanEmb[(size_t)bs * D_ + d] = m;
    }
}

// ---------------------------------------------------------------------------
// Vertex masked mean: one block per (b, v).
// ---------------------------------------------------------------------------
__global__ void __launch_bounds__(256) vertex_mean_kernel(const float* __restrict__ spanEmb,
                                                          const int* __restrict__ vidx,
                                                          const int* __restrict__ vmask,
                                                          float* __restrict__ vertexEmb) {
    const int bv = blockIdx.x;                // b*V + v
    const int b = bv / V_;
    int   idxs[C_];
    float msk[C_];
    float cnt = 0.0f;
    #pragma unroll
    for (int c = 0; c < C_; ++c) {
        idxs[c] = vidx[bv * C_ + c];
        msk[c]  = (float)vmask[bv * C_ + c];
        cnt += msk[c];
    }
    const float inv = 1.0f / (cnt > 1.0f ? cnt : 1.0f);
    for (int d = threadIdx.x; d < D_; d += 256) {
        float s = 0.0f;
        #pragma unroll
        for (int c = 0; c < C_; ++c)
            if (msk[c] != 0.0f)
                s += spanEmb[((size_t)b * NS_ + idxs[c]) * D_ + d];
        vertexEmb[(size_t)bv * D_ + d] = s * inv;
    }
}

// ---------------------------------------------------------------------------
// Build feat row (f16, K padded to 2368): [head | eh | tail | et | head*tail | 0]
// One block per (b, r).
// ---------------------------------------------------------------------------
__global__ void __launch_bounds__(256) build_feat_kernel(const float* __restrict__ vertexEmb,
                                                         const int* __restrict__ hti,
                                                         const int* __restrict__ dht,
                                                         const int* __restrict__ dth,
                                                         const float* __restrict__ dis,
                                                         _Float16* __restrict__ feat) {
    const int br = blockIdx.x;                // b*R + r
    const int b = br / R_;
    const int h = hti[br * 2 + 0];
    const int t = hti[br * 2 + 1];
    const float* hd = vertexEmb + ((size_t)b * V_ + h) * D_;
    const float* tl = vertexEmb + ((size_t)b * V_ + t) * D_;
    const float* eh = dis + (size_t)dht[br] * DIS_;
    const float* et = dis + (size_t)dth[br] * DIS_;
    _Float16* row = feat + (size_t)br * KP1_;
    for (int j = threadIdx.x; j < KP1_; j += 256) {
        float v;
        if      (j < D_)         v = hd[j];
        else if (j < FIN_)       v = eh[j - D_];
        else if (j < FIN_ + D_)  v = tl[j - FIN_];
        else if (j < 2 * FIN_)   v = et[j - FIN_ - D_];
        else if (j < INPROJ_)    v = hd[j - 2 * FIN_] * tl[j - 2 * FIN_];
        else                     v = 0.0f;
        row[j] = (_Float16)v;
    }
}

// ---------------------------------------------------------------------------
// GEMM1 + ReLU: H = relu(feat @ W1)  [8192 x 2368] x [2368 x 384] -> f16 H
// Block tile 128x64, K-step 32, double-buffered LDS filled with CDNA5
// async-to-LDS copies. 8 waves/block; each wave computes a 16x64 strip.
// B fragments are loaded just-in-time per WMMA (low VGPR pressure -> no
// accumulator copies); steady state is branch-free, last K-step peeled.
// ---------------------------------------------------------------------------
__global__ void __launch_bounds__(256) gemm1_relu_wmma_kernel(const _Float16* __restrict__ A,
                                                              const _Float16* __restrict__ BT,
                                                              _Float16* __restrict__ H) {
    __shared__ __align__(16) _Float16 sA[2][BM_ * BK_];   // 2 x 8 KB
    __shared__ __align__(16) _Float16 sB[2][BN_ * BK_];   // 2 x 4 KB

    const int tid  = threadIdx.x;
    const int lane = tid & 31;
    const int wave = tid >> 5;                // 0..7 -> M sub-tile
    const int m0   = blockIdx.x * BM_;
    const int n0   = blockIdx.y * BN_;
    const int mn   = lane & 15;               // M for A-frag, N for B-frag/D
    const int half = lane >> 4;

    // Per-thread staged 16B chunks. A: 2 chunks of 512; B: 1 chunk of 256.
    const int ar0 = tid >> 2,         aq0 = tid & 3;
    const int ar1 = (tid + 256) >> 2, aq1 = (tid + 256) & 3;
    const int brw = tid >> 2,         bq  = tid & 3;
    const _Float16* gA0 = A  + (size_t)(m0 + ar0) * KP1_ + aq0 * 8;
    const _Float16* gA1 = A  + (size_t)(m0 + ar1) * KP1_ + aq1 * 8;
    const _Float16* gB0 = BT + (size_t)(n0 + brw) * KP1_ + bq * 8;

    uint32_t dA0[2], dA1[2], dB0[2];
    #pragma unroll
    for (int p = 0; p < 2; ++p) {
        dA0[p] = (uint32_t)(size_t)&sA[p][ar0 * BK_ + aq0 * 8];
        dA1[p] = (uint32_t)(size_t)&sA[p][ar1 * BK_ + aq1 * 8];
        dB0[p] = (uint32_t)(size_t)&sB[p][brw * BK_ + bq * 8];
    }

    auto stage = [&](int p, int k0) {
        async_ld_b128(dA0[p], (const void*)(gA0 + k0));
        async_ld_b128(dA1[p], (const void*)(gA1 + k0));
        async_ld_b128(dB0[p], (const void*)(gB0 + k0));
    };

    v8f acc[4] = {};

    // A fragment loaded once per K-step; B fragments just-in-time per WMMA.
    auto compute = [&](int p) {
        const _Float16* la = &sA[p][(wave * 16 + mn) * BK_ + half * 8];
        union { v16h v; v8h h[2]; } ua;
        ua.h[0] = *(const v8h*)(la);
        ua.h[1] = *(const v8h*)(la + 16);
        #pragma unroll
        for (int j = 0; j < 4; ++j) {
            const _Float16* lb = &sB[p][(j * 16 + mn) * BK_ + half * 8];
            union { v16h v; v8h h[2]; } ub;
            ub.h[0] = *(const v8h*)(lb);
            ub.h[1] = *(const v8h*)(lb + 16);
            acc[j] = __builtin_amdgcn_wmma_f32_16x16x32_f16(
                         false, ua.v, false, ub.v, (short)0, acc[j], false, false);
        }
    };

    stage(0, 0);
    wait_async0();
    __syncthreads();

    int buf = 0;
    // Steady state: KP1_/BK_ - 1 iterations, branch-free interior.
    for (int k0 = 0; k0 < KP1_ - BK_; k0 += BK_) {
        stage(buf ^ 1, k0 + BK_);   // overlap next stage with this WMMA chain
        compute(buf);
        wait_async0();              // next-stage LDS writes landed
        __syncthreads();            // publish to all waves
        buf ^= 1;
    }
    compute(buf);                   // peeled final K-step

    // Fused ReLU epilogue; D layout: acc[j][r] -> M = half*8 + r, N = j*16 + mn
    #pragma unroll
    for (int j = 0; j < 4; ++j) {
        #pragma unroll
        for (int r = 0; r < 8; ++r) {
            float v = acc[j][r] > 0.0f ? acc[j][r] : 0.0f;
            H[(size_t)(m0 + wave * 16 + half * 8 + r) * HID_ + n0 + j * 16 + mn] = (_Float16)v;
        }
    }
}

// ---------------------------------------------------------------------------
// GEMM2 + bias: out = H @ W2 + b2   [8192 x 384] x [384 x 97(->112)]
// 1 wave per block; 12 WMMA K-steps; W2T (86 KB) is L2-resident.
// ---------------------------------------------------------------------------
__global__ void __launch_bounds__(32) gemm2_bias_wmma_kernel(const _Float16* __restrict__ A,
                                                             const _Float16* __restrict__ BT,
                                                             const float* __restrict__ bias,
                                                             float* __restrict__ out) {
    const int lane = threadIdx.x & 31;
    const int m0 = blockIdx.x * 16;
    const int n0 = blockIdx.y * 16;
    const int mn   = lane & 15;
    const int half = lane >> 4;

    const _Float16* arow = A  + (size_t)(m0 + mn) * HID_ + half * 8;
    const _Float16* bcol = BT + (size_t)(n0 + mn) * HID_ + half * 8;

    v8f c = {};
    #pragma unroll
    for (int k0 = 0; k0 < HID_; k0 += 32) {
        union { v16h v; v8h h[2]; } ua, ub;
        ua.h[0] = *(const v8h*)(arow + k0);
        ua.h[1] = *(const v8h*)(arow + k0 + 16);
        ub.h[0] = *(const v8h*)(bcol + k0);
        ub.h[1] = *(const v8h*)(bcol + k0 + 16);
        c = __builtin_amdgcn_wmma_f32_16x16x32_f16(
                false, ua.v, false, ub.v, (short)0, c, false, false);
    }
    const int n = n0 + mn;
    if (n < REL_) {
        const float bb = bias[n];
        #pragma unroll
        for (int r = 0; r < 8; ++r)
            out[(size_t)(m0 + half * 8 + r) * REL_ + n] = c[r] + bb;
    }
}

// ---------------------------------------------------------------------------
// Host launcher
// ---------------------------------------------------------------------------
extern "C" void kernel_launch(void* const* d_in, const int* in_sizes, int n_in,
                              void* d_out, int out_size, void* d_ws, size_t ws_size,
                              hipStream_t stream) {
    const float* sr    = (const float*)d_in[0];   // sentence_repr (B,S,D)
    const int*   esi   = (const int*)  d_in[1];   // entity_span_indices (B,NS,2)
    // d_in[2] entity_span_indices_mask: unused by reference
    const int*   vidx  = (const int*)  d_in[3];   // vertex_indices (B,V,C)
    const int*   vmask = (const int*)  d_in[4];   // vertex_indices_mask (B,V,C)
    const int*   hti   = (const int*)  d_in[5];   // head_tail_indices (B,R,2)
    // d_in[6] relation_mask: unused by reference
    const int*   dht   = (const int*)  d_in[7];   // dis_h_2_t (B,R)
    const int*   dth   = (const int*)  d_in[8];   // dis_t_2_h (B,R)
    const float* dis   = (const float*)d_in[9];   // dis_embed (20,DIS)
    const float* W1    = (const float*)d_in[10];  // (INPROJ, HID)
    const float* W2    = (const float*)d_in[11];  // (HID, REL)
    const float* b2    = (const float*)d_in[12];  // (REL,)
    float* out = (float*)d_out;                   // (B,R,REL)

    // Workspace carve-out (256B aligned blocks)
    char* ws = (char*)d_ws;
    size_t off = 0;
    auto carve = [&](size_t bytes) -> void* {
        void* p = ws + off;
        off += (bytes + 255) & ~(size_t)255;
        return p;
    };
    float*    spanEmb   = (float*)   carve((size_t)B_ * NS_ * D_ * sizeof(float));
    float*    vertexEmb = (float*)   carve((size_t)B_ * V_ * D_ * sizeof(float));
    _Float16* featF16   = (_Float16*)carve((size_t)M_ * KP1_ * sizeof(_Float16));
    _Float16* W1T       = (_Float16*)carve((size_t)HID_ * KP1_ * sizeof(_Float16));
    _Float16* Hf16      = (_Float16*)carve((size_t)M_ * HID_ * sizeof(_Float16));
    _Float16* W2T       = (_Float16*)carve((size_t)NP2_ * HID_ * sizeof(_Float16));
    (void)ws_size; (void)in_sizes; (void)n_in; (void)out_size;

    // Weight packing (f32 -> transposed, padded f16)
    pack_w1t_kernel<<<(HID_ * KP1_ + 255) / 256, 256, 0, stream>>>(W1, W1T);
    pack_w2t_kernel<<<(NP2_ * HID_ + 255) / 256, 256, 0, stream>>>(W2, W2T);

    // Gather / pooling pipeline
    span_max_kernel   <<<B_ * NS_, 256, 0, stream>>>(sr, esi, spanEmb);
    vertex_mean_kernel<<<B_ * V_,  256, 0, stream>>>(spanEmb, vidx, vmask, vertexEmb);
    build_feat_kernel <<<B_ * R_,  256, 0, stream>>>(vertexEmb, hti, dht, dth, dis, featF16);

    // WMMA GEMMs
    gemm1_relu_wmma_kernel<<<dim3(M_ / BM_, HID_ / BN_), 256, 0, stream>>>(featF16, W1T, Hf16);
    gemm2_bias_wmma_kernel<<<dim3(M_ / 16, NP2_ / 16), 32, 0, stream>>>(Hf16, W2T, b2, out);
}
